// TransformerCrf_4252017623477
// MI455X (gfx1250) — matrix operand order repaired and checked by reading, not compiled, not run on hardware
//
#include <hip/hip_runtime.h>
#include <hip/hip_bf16.h>
#include <math.h>

// ---------------- problem constants ----------------
#define S_LEN   2048
#define H_DIM   512
#define F_DIM   2048
#define T_TAGS  32
#define N_LAYERS 6
#define START_TAG 30
#define STOP_TAG  31
#define NEG_VAL  (-10000.0f)
#define EPS_LN   1e-5f

#define SH (S_LEN * H_DIM)   // 1,048,576
#define SF (S_LEN * F_DIM)   // 4,194,304  (== S*S since F == S)
#define HH (H_DIM * H_DIM)
#define HF (H_DIM * F_DIM)
#define ST (S_LEN * T_TAGS)

// ---------------- vector types ----------------
typedef __attribute__((ext_vector_type(16))) __bf16 v16bf;
typedef __attribute__((ext_vector_type(8)))  __bf16 v8bf;
typedef __attribute__((ext_vector_type(8)))  float  v8f;
typedef __attribute__((ext_vector_type(4)))  unsigned int u32x4;

__device__ __forceinline__ __bf16 f2bf(float f) {
    unsigned u = __builtin_bit_cast(unsigned, f);
    unsigned r = (u + 0x7FFFu + ((u >> 16) & 1u)) >> 16;
    unsigned short h = (unsigned short)r;
    return __builtin_bit_cast(__bf16, h);
}

// ---------------- GEMM ----------------
// C[M,N](fp32, optional) / Cbf[M,N](bf16, optional)
//   = act(A_bf16[M,K] @ Bt_bf16[N,K]^T * scale + bias + resid)
// Double-buffered LDS, software pipelined. K multiple of 32, M multiple of 128.
#define BM 128
#define BN 128
#define BK 32

__global__ __launch_bounds__(256)
void gemm_bf16_wmma(const __bf16* __restrict__ A, const __bf16* __restrict__ Bt,
                    const float* __restrict__ bias, const float* __restrict__ resid,
                    float* __restrict__ C, __bf16* __restrict__ Cbf,
                    int M, int N, int K, float scale, int relu)
{
    __shared__ __bf16 As[2][BM * BK];   // 2 x 8 KB
    __shared__ __bf16 Bs[2][BN * BK];   // 2 x 8 KB

    const int tid  = threadIdx.x;
    const int lane = tid & 31;
    const int wave = tid >> 5;       // 8 waves
    const int wm   = wave & 1;       // 2 waves along M (64 rows each)
    const int wn   = wave >> 1;      // 4 waves along N (32 cols each)
    const int bm   = blockIdx.y * BM;
    const int bn   = blockIdx.x * BN;
    const bool fullN = (bn + BN <= N);

    // staging: each thread owns 32 contiguous bytes of one tile row
    const int srow = tid >> 1;            // 0..127
    const int scol = (tid & 1) * 16;      // 0 or 16 (elements)

    const int r15   = lane & 15;
    const int khalf = lane >> 4;

    v8f acc[4][2];
    for (int i = 0; i < 4; ++i)
        for (int j = 0; j < 2; ++j)
            acc[i][j] = v8f{0.f,0.f,0.f,0.f,0.f,0.f,0.f,0.f};

    const __bf16* Arow = A  + (size_t)(bm + srow) * K + scol;
    const __bf16* Brow = Bt + (size_t)(bn + srow) * K + scol;
    const bool bok = fullN || (bn + srow < N);

    u32x4 ra0, ra1, rb0, rb1;
    // prologue: fetch stage 0
    ra0 = *(const u32x4*)(Arow);
    ra1 = *(const u32x4*)(Arow + 8);
    rb0 = u32x4{};
    rb1 = u32x4{};
    if (bok) { rb0 = *(const u32x4*)(Brow); rb1 = *(const u32x4*)(Brow + 8); }
    *(u32x4*)(&As[0][srow * BK + scol])     = ra0;
    *(u32x4*)(&As[0][srow * BK + scol + 8]) = ra1;
    *(u32x4*)(&Bs[0][srow * BK + scol])     = rb0;
    *(u32x4*)(&Bs[0][srow * BK + scol + 8]) = rb1;
    __syncthreads();

    const int nst = K / BK;
    for (int s = 0; s < nst; ++s) {
        const int buf = s & 1;
        const bool more = (s + 1) < nst;

        // issue next stage's global loads early (overlaps with WMMA below)
        if (more) {
            const int k0 = (s + 1) * BK;
            ra0 = *(const u32x4*)(Arow + k0);
            ra1 = *(const u32x4*)(Arow + k0 + 8);
            if (fullN) {
                rb0 = *(const u32x4*)(Brow + k0);
                rb1 = *(const u32x4*)(Brow + k0 + 8);
            } else {
                rb0 = u32x4{}; rb1 = u32x4{};
                if (bok) { rb0 = *(const u32x4*)(Brow + k0); rb1 = *(const u32x4*)(Brow + k0 + 8); }
            }
        }

        // ---- fragments from LDS[buf] ----
        // A (ISA 16-bit A layout): lanes 0-15 K[0..7]+K[16..23], lanes 16-31 K[8..15]+K[24..31]
        v16bf afrag[4];
        for (int i = 0; i < 4; ++i) {
            int row = wm * 64 + i * 16 + r15;
            v8bf lo = *(const v8bf*)(&As[buf][row * BK + khalf * 8]);
            v8bf hi = *(const v8bf*)(&As[buf][row * BK + khalf * 8 + 16]);
            for (int e = 0; e < 8; ++e) { afrag[i][e] = lo[e]; afrag[i][e + 8] = hi[e]; }
        }
        // B: lane = col (r15), same K layout as A: lanes 0-15 K[0..7]+K[16..23], lanes 16-31 K[8..15]+K[24..31]
        v16bf bfrag[2];
        for (int j = 0; j < 2; ++j) {
            int row = wn * 32 + j * 16 + r15;
            v8bf lo = *(const v8bf*)(&Bs[buf][row * BK + khalf * 8]);
            v8bf hi = *(const v8bf*)(&Bs[buf][row * BK + khalf * 8 + 16]);
            for (int e = 0; e < 8; ++e) { bfrag[j][e] = lo[e]; bfrag[j][e + 8] = hi[e]; }
        }

        for (int i = 0; i < 4; ++i)
            for (int j = 0; j < 2; ++j)
                acc[i][j] = __builtin_amdgcn_wmma_f32_16x16x32_bf16(
                    false, afrag[i], false, bfrag[j], (short)0, acc[i][j], false, false);

        // stage fetched registers into the other buffer
        if (more) {
            const int nb = buf ^ 1;
            *(u32x4*)(&As[nb][srow * BK + scol])     = ra0;
            *(u32x4*)(&As[nb][srow * BK + scol + 8]) = ra1;
            *(u32x4*)(&Bs[nb][srow * BK + scol])     = rb0;
            *(u32x4*)(&Bs[nb][srow * BK + scol + 8]) = rb1;
        }
        __syncthreads();
    }

    // ---- epilogue: lane = col (r15), VGPR e = row e (+8 for lanes 16-31) ----
    for (int i = 0; i < 4; ++i) {
        for (int j = 0; j < 2; ++j) {
            int col = bn + wn * 32 + j * 16 + r15;
            if (col >= N) continue;
            float bv = bias ? bias[col] : 0.0f;
            for (int e = 0; e < 8; ++e) {
                int row = bm + wm * 64 + i * 16 + e + khalf * 8;
                float val = acc[i][j][e] * scale + bv;
                if (resid) val += resid[(size_t)row * N + col];
                if (relu)  val = fmaxf(val, 0.0f);
                if (C)     C[(size_t)row * N + col] = val;
                if (Cbf)   Cbf[(size_t)row * N + col] = f2bf(val);
            }
        }
    }
}

// ---------------- converters ----------------
__global__ void f32_to_bf16_k(const float* __restrict__ src, __bf16* __restrict__ dst, int n) {
    int i = blockIdx.x * blockDim.x + threadIdx.x;
    if (i < n) dst[i] = f2bf(src[i]);
}

// dst[C][R] = (bf16) src[R][C]
__global__ void f32_to_bf16_T_k(const float* __restrict__ src, __bf16* __restrict__ dst, int R, int Cc) {
    __shared__ float tile[32][33];
    int c0 = blockIdx.x * 32, r0 = blockIdx.y * 32;
    int tx = threadIdx.x, ty = threadIdx.y;   // block (32,8)
    for (int i = 0; i < 32; i += 8) {
        int r = r0 + ty + i, c = c0 + tx;
        tile[ty + i][tx] = (r < R && c < Cc) ? src[(size_t)r * Cc + c] : 0.f;
    }
    __syncthreads();
    for (int i = 0; i < 32; i += 8) {
        int c = c0 + ty + i, r = r0 + tx;
        if (c < Cc && r < R) dst[(size_t)c * R + r] = f2bf(tile[tx][ty + i]);
    }
}

__global__ void copy_f32_k(const float* __restrict__ src, float* __restrict__ dst, int n) {
    int i = blockIdx.x * blockDim.x + threadIdx.x;
    if (i < n) dst[i] = src[i];
}

// ---------------- embedding gather ----------------
__global__ __launch_bounds__(256)
void embed_gather_k(const int* __restrict__ sent, const float* __restrict__ embed, float* __restrict__ x) {
    int srow = blockIdx.x;
    int tok  = sent[srow];
    for (int i = threadIdx.x; i < H_DIM; i += 256)
        x[(size_t)srow * H_DIM + i] = embed[(size_t)tok * H_DIM + i];
}

// ---------------- row softmax: fp32 in-place + bf16 copy ----------------
__global__ __launch_bounds__(256)
void softmax_row_k(float* __restrict__ x, __bf16* __restrict__ obf, int n) {
    __shared__ float red[8];
    __shared__ float stat[2];
    const int row = blockIdx.x;
    float* xr = x + (size_t)row * n;
    const int tid = threadIdx.x, lane = tid & 31, wv = tid >> 5;

    float m = -3.4e38f;
    for (int i = tid; i < n; i += 256) m = fmaxf(m, xr[i]);
    for (int o = 16; o > 0; o >>= 1) m = fmaxf(m, __shfl_down(m, o));
    if (lane == 0) red[wv] = m;
    __syncthreads();
    if (tid == 0) { float t = red[0]; for (int i = 1; i < 8; ++i) t = fmaxf(t, red[i]); stat[0] = t; }
    __syncthreads();
    const float mx = stat[0];

    float s = 0.f;
    for (int i = tid; i < n; i += 256) { float e = __expf(xr[i] - mx); xr[i] = e; s += e; }
    for (int o = 16; o > 0; o >>= 1) s += __shfl_down(s, o);
    if (lane == 0) red[wv] = s;
    __syncthreads();
    if (tid == 0) { float t = 0.f; for (int i = 0; i < 8; ++i) t += red[i]; stat[1] = t; }
    __syncthreads();
    const float inv = 1.0f / stat[1];
    __bf16* orow = obf + (size_t)row * n;
    for (int i = tid; i < n; i += 256) { float v = xr[i] * inv; xr[i] = v; orow[i] = f2bf(v); }
}

// ---------------- row layernorm: fp32 out + optional bf16 out ----------------
__global__ __launch_bounds__(256)
void layernorm_row_k(const float* __restrict__ x, const float* __restrict__ w,
                     const float* __restrict__ b, float* __restrict__ out,
                     __bf16* __restrict__ obf, int n) {
    __shared__ float red[8];
    __shared__ float stat[2];
    const int row = blockIdx.x;
    const float* xr = x + (size_t)row * n;
    const int tid = threadIdx.x, lane = tid & 31, wv = tid >> 5;

    float s = 0.f;
    for (int i = tid; i < n; i += 256) s += xr[i];
    for (int o = 16; o > 0; o >>= 1) s += __shfl_down(s, o);
    if (lane == 0) red[wv] = s;
    __syncthreads();
    if (tid == 0) { float t = 0.f; for (int i = 0; i < 8; ++i) t += red[i]; stat[0] = t / n; }
    __syncthreads();
    const float mean = stat[0];

    float v = 0.f;
    for (int i = tid; i < n; i += 256) { float d = xr[i] - mean; v += d * d; }
    for (int o = 16; o > 0; o >>= 1) v += __shfl_down(v, o);
    if (lane == 0) red[wv] = v;
    __syncthreads();
    if (tid == 0) { float t = 0.f; for (int i = 0; i < 8; ++i) t += red[i]; stat[1] = t / n; }
    __syncthreads();
    const float inv = rsqrtf(stat[1] + EPS_LN);
    for (int i = tid; i < n; i += 256) {
        float val = (xr[i] - mean) * inv * w[i] + b[i];
        out[(size_t)row * n + i] = val;
        if (obf) obf[(size_t)row * n + i] = f2bf(val);
    }
}

// ---------------- Viterbi: one wave32, lane = prev tag ----------------
__global__ void viterbi_k(const float* __restrict__ feats, const float* __restrict__ trans,
                          int* __restrict__ bptrs, float* __restrict__ out) {
    const int lane = threadIdx.x;  // 0..31
    __shared__ float tsh[T_TAGS * T_TAGS];
    for (int i = lane; i < T_TAGS * T_TAGS; i += 32) tsh[i] = trans[i];
    __syncthreads();

    float fv = (lane == START_TAG) ? 0.f : NEG_VAL;
    for (int t = 0; t < S_LEN; ++t) {
        float myfeat = feats[t * T_TAGS + lane];
        float nfv = NEG_VAL;
        int   bp  = 0;
        for (int nxt = 0; nxt < T_TAGS; ++nxt) {
            float val = fv + tsh[nxt * T_TAGS + lane];
            int   idx = lane;
            for (int o = 16; o > 0; o >>= 1) {
                float ov = __shfl_down(val, o);
                int   oi = __shfl_down(idx, o);
                if (ov > val || (ov == val && oi < idx)) { val = ov; idx = oi; }
            }
            val = __shfl(val, 0);
            idx = __shfl(idx, 0);
            if (lane == nxt) { nfv = val + myfeat; bp = idx; }
        }
        fv = nfv;
        bptrs[t * T_TAGS + lane] = bp;
    }
    float val = fv + tsh[STOP_TAG * T_TAGS + lane];
    int   idx = lane;
    for (int o = 16; o > 0; o >>= 1) {
        float ov = __shfl_down(val, o);
        int   oi = __shfl_down(idx, o);
        if (ov > val || (ov == val && oi < idx)) { val = ov; idx = oi; }
    }
    val = __shfl(val, 0);
    idx = __shfl(idx, 0);
    if (lane == 0) {
        out[0] = val;
        int tag = idx;
        for (int t = S_LEN - 1; t >= 0; --t) {
            out[1 + t] = (float)tag;
            tag = bptrs[t * T_TAGS + tag];
        }
    }
}

// ---------------- orchestration ----------------
extern "C" void kernel_launch(void* const* d_in, const int* in_sizes, int n_in,
                              void* d_out, int out_size, void* d_ws, size_t ws_size,
                              hipStream_t stream)
{
    (void)in_sizes; (void)n_in; (void)out_size; (void)ws_size;

    const int*   sentence = (const int*)  d_in[0];
    const float* embed    = (const float*)d_in[1];
    const float* eaw  = (const float*)d_in[2];
    const float* eab  = (const float*)d_in[3];
    const float* ew1  = (const float*)d_in[4];
    const float* eb1  = (const float*)d_in[5];
    const float* ew2  = (const float*)d_in[6];
    const float* eb2  = (const float*)d_in[7];
    const float* elnw = (const float*)d_in[8];
    const float* elnb = (const float*)d_in[9];
    const float* daw  = (const float*)d_in[10];
    const float* dab  = (const float*)d_in[11];
    const float* dw1  = (const float*)d_in[12];
    const float* db1  = (const float*)d_in[13];
    const float* dw2  = (const float*)d_in[14];
    const float* db2  = (const float*)d_in[15];
    const float* dlnw = (const float*)d_in[16];
    const float* dlnb = (const float*)d_in[17];
    const float* enw  = (const float*)d_in[18];
    const float* enb  = (const float*)d_in[19];
    const float* dnw  = (const float*)d_in[20];
    const float* dnb  = (const float*)d_in[21];
    const float* h2tw = (const float*)d_in[22];
    const float* h2tb = (const float*)d_in[23];
    const float* trans = (const float*)d_in[24];

    // workspace carve-up
    char* wsb = (char*)d_ws;
    size_t off = 0;
    auto carve = [&](size_t bytes) -> char* {
        char* p = wsb + off;
        off = (off + bytes + 255) & ~(size_t)255;
        return p;
    };
    float*  xF    = (float*) carve((size_t)SH * 4);
    float*  hF    = (float*) carve((size_t)SH * 4);
    float*  yF    = (float*) carve((size_t)SH * 4);
    float*  memF  = (float*) carve((size_t)SH * 4);
    float*  vF    = (float*) carve((size_t)SH * 4);
    float*  aF    = (float*) carve((size_t)SH * 4);
    float*  sF    = (float*) carve((size_t)SF * 4);      // scores [S,S] fp32
    float*  featF = (float*) carve((size_t)ST * 4);
    __bf16* hb    = (__bf16*)carve((size_t)SH * 2);      // bf16 hidden (encoder)
    __bf16* yb    = (__bf16*)carve((size_t)SH * 2);      // bf16 hidden (decoder)
    __bf16* memb  = (__bf16*)carve((size_t)SH * 2);
    __bf16* qb    = (__bf16*)carve((size_t)SH * 2);
    __bf16* kb    = (__bf16*)carve((size_t)SH * 2);
    __bf16* ctxb  = (__bf16*)carve((size_t)SH * 2);
    __bf16* pb    = (__bf16*)carve((size_t)SF * 2);      // softmax probs bf16
    __bf16* ffb   = (__bf16*)carve((size_t)SF * 2);      // relu(ffn1) bf16
    __bf16* Bb    = (__bf16*)carve((size_t)HF * 2);      // transposed weight / V^T
    int*    bptr  = (int*)   carve((size_t)ST * 4);

    const float attn_scale = 0.04419417382f;  // 1/sqrt(512)

    auto conv = [&](const float* src, __bf16* dst, int n) {
        f32_to_bf16_k<<<(n + 255) / 256, 256, 0, stream>>>(src, dst, n);
    };
    auto convT = [&](const float* src, __bf16* dst, int R, int Cc) {
        dim3 g((Cc + 31) / 32, (R + 31) / 32), bl(32, 8);
        f32_to_bf16_T_k<<<g, bl, 0, stream>>>(src, dst, R, Cc);
    };
    auto gemm = [&](const __bf16* A, const __bf16* Bt, const float* bias, const float* resid,
                    float* Cf, __bf16* Cbf, int M, int N, int K, float scale, int relu) {
        dim3 g((N + BN - 1) / BN, (M + BM - 1) / BM);
        gemm_bf16_wmma<<<g, 256, 0, stream>>>(A, Bt, bias, resid, Cf, Cbf, M, N, K, scale, relu);
    };
    auto ln = [&](const float* x, const float* w, const float* b, float* out, __bf16* obf) {
        layernorm_row_k<<<S_LEN, 256, 0, stream>>>(x, w, b, out, obf, H_DIM);
    };

    // attention with bf16 A-operands; writes fp32 out (with residual added)
    auto attention = [&](const __bf16* qin_bf, const __bf16* kvin_bf,
                         const float* W, const float* B, const float* resid, float* outF) {
        convT(W + 0 * HH, Bb, H_DIM, H_DIM);
        gemm(qin_bf, Bb, B + 0 * H_DIM, nullptr, nullptr, qb, S_LEN, H_DIM, H_DIM, 1.f, 0);
        convT(W + 1 * HH, Bb, H_DIM, H_DIM);
        gemm(kvin_bf, Bb, B + 1 * H_DIM, nullptr, nullptr, kb, S_LEN, H_DIM, H_DIM, 1.f, 0);
        convT(W + 2 * HH, Bb, H_DIM, H_DIM);
        gemm(kvin_bf, Bb, B + 2 * H_DIM, nullptr, vF, nullptr, S_LEN, H_DIM, H_DIM, 1.f, 0);
        // scores = q k^T / sqrt(H)  (k already [N=S, K=H])
        gemm(qb, kb, nullptr, nullptr, sF, nullptr, S_LEN, S_LEN, H_DIM, attn_scale, 0);
        softmax_row_k<<<S_LEN, 256, 0, stream>>>(sF, pb, S_LEN);
        // ctx = p v   (v transposed to [H, S])
        convT(vF, Bb, S_LEN, H_DIM);
        gemm(pb, Bb, nullptr, nullptr, nullptr, ctxb, S_LEN, H_DIM, S_LEN, 1.f, 0);
        // out projection + residual
        convT(W + 3 * HH, Bb, H_DIM, H_DIM);
        gemm(ctxb, Bb, B + 3 * H_DIM, resid, outF, nullptr, S_LEN, H_DIM, H_DIM, 1.f, 0);
    };

    auto ffn = [&](const __bf16* hin_bf, const float* hin_resid,
                   const float* W1, const float* B1, const float* W2, const float* B2, float* outF) {
        convT(W1, Bb, H_DIM, F_DIM);                                          // -> [F, H]
        gemm(hin_bf, Bb, B1, nullptr, nullptr, ffb, S_LEN, F_DIM, H_DIM, 1.f, 1); // ReLU, bf16 only
        convT(W2, Bb, F_DIM, H_DIM);                                          // -> [H, F]
        gemm(ffb, Bb, B2, hin_resid, outF, nullptr, S_LEN, H_DIM, F_DIM, 1.f, 0);
    };

    // ---- embeddings ----
    embed_gather_k<<<S_LEN, 256, 0, stream>>>(sentence, embed, xF);
    copy_f32_k<<<(SH + 255) / 256, 256, 0, stream>>>(xF, hF, SH);
    conv(xF, hb, SH);

    // ---- encoder ----
    for (int i = 0; i < N_LAYERS; ++i) {
        attention(hb, hb, eaw + (size_t)i * 4 * HH, eab + (size_t)i * 4 * H_DIM, hF, aF);
        ln(aF, elnw + (size_t)i * 2 * H_DIM, elnb + (size_t)i * 2 * H_DIM, hF, hb);
        ffn(hb, hF, ew1 + (size_t)i * HF, eb1 + (size_t)i * F_DIM,
            ew2 + (size_t)i * HF, eb2 + (size_t)i * H_DIM, aF);
        ln(aF, elnw + (size_t)i * 2 * H_DIM + H_DIM, elnb + (size_t)i * 2 * H_DIM + H_DIM, hF, hb);
    }
    ln(hF, enw, enb, memF, memb);

    // ---- decoder ----
    copy_f32_k<<<(SH + 255) / 256, 256, 0, stream>>>(xF, yF, SH);
    conv(xF, yb, SH);
    for (int i = 0; i < N_LAYERS; ++i) {
        const float* W = daw + (size_t)i * 8 * HH;
        const float* B = dab + (size_t)i * 8 * H_DIM;
        attention(yb, yb, W, B, yF, aF);                            // self
        ln(aF, dlnw + (size_t)i * 3 * H_DIM, dlnb + (size_t)i * 3 * H_DIM, yF, yb);
        attention(yb, memb, W + 4 * HH, B + 4 * H_DIM, yF, aF);     // cross
        ln(aF, dlnw + (size_t)i * 3 * H_DIM + H_DIM, dlnb + (size_t)i * 3 * H_DIM + H_DIM, yF, yb);
        ffn(yb, yF, dw1 + (size_t)i * HF, db1 + (size_t)i * F_DIM,
            dw2 + (size_t)i * HF, db2 + (size_t)i * H_DIM, aF);
        ln(aF, dlnw + (size_t)i * 3 * H_DIM + 2 * H_DIM, dlnb + (size_t)i * 3 * H_DIM + 2 * H_DIM, yF, yb);
    }
    ln(yF, dnw, dnb, aF, yb);   // final decoder norm -> yb (bf16)

    // ---- features = y @ h2t_w + h2t_b ----
    convT(h2tw, Bb, H_DIM, T_TAGS);                                 // -> [T, H]
    gemm(yb, Bb, h2tb, nullptr, featF, nullptr, S_LEN, T_TAGS, H_DIM, 1.f, 0);

    // ---- Viterbi decode ----
    viterbi_k<<<1, 32, 0, stream>>>(featF, trans, bptr, (float*)d_out);
}
